// Encoder_23141283791591
// MI455X (gfx1250) — 8-bit matrix operand order repaired and checked by reading, compiled, not run on hardware
//
#include <hip/hip_runtime.h>
#include <hip/hip_bf16.h>

typedef __attribute__((ext_vector_type(8))) int v8i;
typedef __attribute__((ext_vector_type(4))) int v4i;

#if defined(__has_builtin)
#if __has_builtin(__builtin_amdgcn_global_load_async_to_lds_b32) && \
    __has_builtin(__builtin_amdgcn_s_wait_asynccnt)
#define ASYNC_LDS 1
#endif
#endif
#ifndef ASYNC_LDS
#define ASYNC_LDS 0
#endif

#if ASYNC_LDS
typedef __attribute__((address_space(1))) int g_int;
typedef __attribute__((address_space(3))) int l_int;
#endif

__global__ void pack_w_kernel(const float* __restrict__ w, unsigned char* __restrict__ out,
                              int Cout, int Cin, int KK, int CP, int nKc, int nCb16)
{
    int i = blockIdx.x * blockDim.x + threadIdx.x;
    int total = nCb16 * nKc * 1024;
    if (i >= total) return;
    int byte = i & 3;
    int v    = (i >> 2) & 7;
    int lane = (i >> 5) & 31;
    int blk  = i >> 10;
    int kc   = blk % nKc;
    int cb16 = blk / nKc;
    int cout = cb16 * 16 + (lane & 15);
    int k = kc * 64 + ((lane >> 4) << 3)
          + ((v & 1) << 2) + (((v >> 1) & 1) << 4) + (((v >> 2) & 1) << 5) + byte;
    int c  = k % CP;
    int p  = k / CP;
    int KHKW = KK * KK;
    int kh = p / KK, kw = p - (p / KK) * KK;
    signed char val = 0;
    if (cout < Cout && c < Cin && p < KHKW)
        val = (signed char)(int)w[(((size_t)cout * Cin + c) * KHKW) + kh * KK + kw];
    out[i] = (unsigned char)val;
}

__global__ void quant_x_nhwc_kernel(const float* __restrict__ x, unsigned char* __restrict__ q,
                                    int npix  )
{
    int i = blockIdx.x * blockDim.x + threadIdx.x;
    if (i >= npix) return;
    const int hw = 256 * 256;
    int b = i / hw, r = i - b * hw;
    unsigned int d = 0;
    #pragma unroll
    for (int c = 0; c < 3; ++c)
        d |= ((unsigned int)(unsigned char)(int)x[((size_t)b * 3 + c) * hw + r]) << (8 * c);
    *(unsigned int*)(q + (size_t)i * 4) = d;
}

template<int CP, int HIN, int WIN, int COUT, int HOUT, int WOUT,
         int KK, int STRIDE, int PAD, int NT, int MODE, int S1, int S2>
__global__ void __launch_bounds__(256)
conv_wmma_kernel(const unsigned char* __restrict__ actIn,
                 const unsigned char* __restrict__ wPack,
                 const float* __restrict__ bias,
                 const float* __restrict__ mul,
                 const float* __restrict__ clipArr, int clipIdx,
                 const float* __restrict__ sclArr,  int sclIdx,
                 unsigned char* __restrict__ actOut,
                 float* __restrict__ outF)
{
    constexpr int KHKW   = KK * KK;
    constexpr int NKC    = (KHKW * CP + 63) / 64;
    constexpr int NCBBLK = (COUT + 127) / 128;
    constexpr int PTOT   = HOUT * WOUT;
    constexpr int TILESP = PTOT / (16 * NT);

    __shared__ __align__(16) unsigned int ldsB[2][NT * 256];

    const int tid  = threadIdx.x;
    const int wave = tid >> 5;
    const int lane = tid & 31;

    int idx = blockIdx.x;
    const int tp    = idx % TILESP; idx /= TILESP;
    const int cbBlk = idx % NCBBLK; idx /= NCBBLK;
    const int b     = idx;
    const int p0    = tp * (16 * NT);

    const int bl    = tid >> 3;
    const int v     = tid & 7;
    const int n     = bl & 15;
    const int klb   = (bl >> 4) << 3;
    const int kOffV = ((v >> 2) << 5) + klb + (((v >> 1) & 1) << 4) + ((v & 1) << 2);
    const unsigned char* __restrict__ inB = actIn + (size_t)b * HIN * WIN * CP;

    int ihB[NT], iwB[NT];
    #pragma unroll
    for (int t = 0; t < NT; ++t) {
        const int pos = p0 + t * 16 + n;
        const int oh = pos / WOUT, ow = pos - (pos / WOUT) * WOUT;
        ihB[t] = oh * STRIDE - PAD;
        iwB[t] = ow * STRIDE - PAD;
    }

    auto buildB = [&](int kc, unsigned int* buf) {
        const int k0 = kc * 64 + kOffV;
        const int c0 = k0 % CP;
        const int p  = k0 / CP;
        const int kh = p / KK, kw = p - (p / KK) * KK;
        #pragma unroll
        for (int t = 0; t < NT; ++t) {
            const int ih = ihB[t] + kh, iw = iwB[t] + kw;
            const bool ok = (p < KHKW) && ((unsigned)ih < (unsigned)HIN)
                                       && ((unsigned)iw < (unsigned)WIN);
            unsigned int* dst = &buf[t * 256 + bl * 8 + v];
#if ASYNC_LDS
            if (ok)
                __builtin_amdgcn_global_load_async_to_lds_b32(
                    (g_int*)(inB + ((size_t)(ih * WIN + iw)) * CP + c0),
                    (l_int*)dst,
                    0, 0);
            else
                *dst = 0;
#else
            unsigned int d = 0;
            if (ok) d = *(const unsigned int*)(inB + ((size_t)(ih * WIN + iw)) * CP + c0);
            *dst = d;
#endif
        }
    };

    const unsigned char* __restrict__ aBase =
        wPack + ((size_t)(cbBlk * 8 + wave) * NKC) * 1024 + (size_t)lane * 32;

    v8i acc[NT];
    #pragma unroll
    for (int t = 0; t < NT; ++t) acc[t] = (v8i){};

    buildB(0, ldsB[0]);
#if ASYNC_LDS
    __builtin_amdgcn_s_wait_asynccnt(0);
#endif
    __syncthreads();

    for (int kc = 0; kc < NKC; ++kc) {
        if (kc + 1 < NKC) buildB(kc + 1, ldsB[(kc + 1) & 1]);

        v8i afrag;
        const v4i* ap = (const v4i*)(aBase + (size_t)kc * 1024);
        ((v4i*)&afrag)[0] = ap[0];
        ((v4i*)&afrag)[1] = ap[1];
        if (kc + 1 < NKC)
            __builtin_prefetch(aBase + (size_t)(kc + 1) * 1024, 0, 0);

        const unsigned int* buf = ldsB[kc & 1];
        #pragma unroll
        for (int t = 0; t < NT; ++t) {
            v8i bfrag;
            const v4i* bp = (const v4i*)&buf[t * 256 + lane * 8];
            ((v4i*)&bfrag)[0] = bp[0];
            ((v4i*)&bfrag)[1] = bp[1];
            acc[t] = __builtin_amdgcn_wmma_i32_16x16x64_iu8(true, afrag, false, bfrag,
                                                            acc[t], false, false);
        }
#if ASYNC_LDS
        __builtin_amdgcn_s_wait_asynccnt(0);
#endif
        __syncthreads();
    }

    const int coutW = (cbBlk * 8 + wave) * 16;
    if (coutW >= COUT) return;

    long long clipI = 0, sclI = 0;
    if constexpr (MODE == 0) {
        clipI = (long long)clipArr[clipIdx];
        sclI  = (long long)sclArr[sclIdx];
    }
    const int nCol     = lane & 15;
    const int mrowBase = (lane >> 4) * 8;
    const int cout0    = coutW + mrowBase;
    constexpr long long HALF1 = 1LL << (S1 - 1);

    long long biasI[8], mulI[8];
    #pragma unroll
    for (int r = 0; r < 8; ++r) {
        biasI[r] = (long long)bias[cout0 + r];
        mulI[r]  = (long long)mul[cout0 + r];
    }

    #pragma unroll
    for (int t = 0; t < NT; ++t) {
        const int pos = p0 + t * 16 + nCol;
        unsigned long long packed = 0;
        #pragma unroll
        for (int r = 0; r < 8; ++r) {
            const long long tt = ((long long)acc[t][r] + biasI[r]) * mulI[r];
            const long long u  = (tt + HALF1) >> S1;
            if constexpr (MODE == 0) {
                long long uc = u < 0 ? 0 : (u > clipI ? clipI : u);
                long long vq = (uc * sclI + (1LL << (S2 - 1))) >> S2;
                packed |= (unsigned long long)(unsigned char)vq << (8 * r);
                if (outF)
                    outF[((size_t)b * COUT + cout0 + r) * (size_t)PTOT + pos] = (float)vq;
            } else if constexpr (MODE == 1) {
                outF[((size_t)b * COUT + cout0 + r) * (size_t)PTOT + pos] = (float)((u + 8) >> 4);
                long long a = u < 0 ? -u : u;
                if (a > 254) a = 254;
                packed |= (unsigned long long)(unsigned char)a << (8 * r);
            } else {
                outF[((size_t)b * COUT + cout0 + r) * (size_t)PTOT + pos] = (float)u;
            }
        }
        if constexpr (MODE != 2)
            *(unsigned long long*)(actOut + ((size_t)b * PTOT + pos) * COUT + cout0) = packed;
    }
}

extern "C" void kernel_launch(void* const* d_in, const int* in_sizes, int n_in,
                              void* d_out, int out_size, void* d_ws, size_t ws_size,
                              hipStream_t stream)
{
    (void)in_sizes; (void)n_in; (void)out_size; (void)ws_size;
    const int NB = 16;
    const float* x     = (const float*)d_in[0];
    const float* w[7]  = { (const float*)d_in[1], (const float*)d_in[3], (const float*)d_in[5],
                           (const float*)d_in[7], (const float*)d_in[9], (const float*)d_in[11],
                           (const float*)d_in[13] };
    const float* bb[7] = { (const float*)d_in[2], (const float*)d_in[4], (const float*)d_in[6],
                           (const float*)d_in[8], (const float*)d_in[10], (const float*)d_in[12],
                           (const float*)d_in[14] };
    const float* mu[7] = { (const float*)d_in[15], (const float*)d_in[16], (const float*)d_in[17],
                           (const float*)d_in[18], (const float*)d_in[21], (const float*)d_in[22],
                           (const float*)d_in[23] };
    const float* clpE  = (const float*)d_in[19];
    const float* sclE  = (const float*)d_in[20];
    const float* clppE = (const float*)d_in[24];
    const float* sclpE = (const float*)d_in[25];
    float* out = (float*)d_out;

    const int Cout[7]  = {128, 128, 128, 192, 128, 128, 128};
    const int Cin[7]   = {3, 128, 128, 128, 192, 128, 128};
    const int CPv[7]   = {4, 128, 128, 128, 192, 128, 128};
    const int KKv[7]   = {5, 5, 5, 5, 3, 5, 5};
    const int nKc[7]   = {2, 50, 50, 50, 27, 50, 50};
    const int nCb16[7] = {8, 8, 8, 16, 8, 8, 8};
    const size_t actBytes[7] = {
        (size_t)NB*128*128*128, (size_t)NB*128*64*64, (size_t)NB*128*32*32,
        (size_t)NB*192*16*16,   (size_t)NB*128*16*16, (size_t)NB*128*8*8,
        (size_t)NB*128*4*4 };

    unsigned char* ws = (unsigned char*)d_ws;
    size_t off = 0;
    auto take = [&](size_t bytes) { size_t o = off; off = (off + bytes + 255) & ~(size_t)255; return o; };
    const size_t oX = take((size_t)NB * 256 * 256 * 4);
    size_t oA[7], oW[7];
    for (int l = 0; l < 7; ++l) oA[l] = take(actBytes[l]);
    for (int l = 0; l < 7; ++l) oW[l] = take((size_t)nCb16[l] * nKc[l] * 1024);

    const size_t oE2 = 0;
    const size_t oE3 = oE2 + (size_t)NB * 128 * 64 * 64;
    const size_t oY  = oE3 + (size_t)NB * 128 * 32 * 32;
    const size_t oP1 = oY  + (size_t)NB * 192 * 16 * 16;
    const size_t oP2 = oP1 + (size_t)NB * 128 * 16 * 16;
    const size_t oZ  = oP2 + (size_t)NB * 128 * 8 * 8;

    {
        int npix = NB * 256 * 256;
        quant_x_nhwc_kernel<<<(npix + 255) / 256, 256, 0, stream>>>(x, ws + oX, npix);
    }
    for (int l = 0; l < 7; ++l) {
        int total = nCb16[l] * nKc[l] * 1024;
        pack_w_kernel<<<(total + 255) / 256, 256, 0, stream>>>(
            w[l], ws + oW[l], Cout[l], Cin[l], KKv[l], CPv[l], nKc[l], nCb16[l]);
    }
    conv_wmma_kernel<4,256,256,128,128,128,5,2,2,8,0,20,19><<<16*1*128,256,0,stream>>>(
        ws+oX, ws+oW[0], bb[0], mu[0], clpE,0, sclE,0, ws+oA[0], nullptr);
    conv_wmma_kernel<128,128,128,128,64,64,5,2,2,8,0,16,19><<<16*1*32,256,0,stream>>>(
        ws+oA[0], ws+oW[1], bb[1], mu[1], clpE,1, sclE,1, ws+oA[1], out+oE2);
    conv_wmma_kernel<128,64,64,128,32,32,5,2,2,8,0,16,19><<<16*1*8,256,0,stream>>>(
        ws+oA[1], ws+oW[2], bb[2], mu[2], clpE,2, sclE,2, ws+oA[2], out+oE3);
    conv_wmma_kernel<128,32,32,192,16,16,5,2,2,8,1,15,4><<<16*2*2,256,0,stream>>>(
        ws+oA[2], ws+oW[3], bb[3], mu[3], nullptr,0, nullptr,0, ws+oA[3], out+oY);
    conv_wmma_kernel<192,16,16,128,16,16,3,1,1,8,0,11,21><<<16*1*2,256,0,stream>>>(
        ws+oA[3], ws+oW[4], bb[4], mu[4], clppE,0, sclpE,0, ws+oA[4], out+oP1);
    conv_wmma_kernel<128,16,16,128,8,8,5,2,2,4,0,14,21><<<16*1*1,256,0,stream>>>(
        ws+oA[4], ws+oW[5], bb[5], mu[5], clppE,1, sclpE,1, ws+oA[5], out+oP2);
    conv_wmma_kernel<128,8,8,128,4,4,5,2,2,1,2,23,1><<<16*1*1,256,0,stream>>>(
        ws+oA[5], ws+oW[6], bb[6], mu[6], nullptr,0, nullptr,0, nullptr, out+oZ);
}
